// CompleteModel_31662498906598
// MI455X (gfx1250) — compile-verified, matrix operand order checked by reading, not run on hardware
//
#include <hip/hip_runtime.h>
#include <hip/hip_bf16.h>

typedef __attribute__((ext_vector_type(16))) __bf16 v16bf;
typedef __attribute__((ext_vector_type(8)))  __bf16 v8bf;
typedef __attribute__((ext_vector_type(4)))  __bf16 v4bf;
typedef __attribute__((ext_vector_type(8)))  float  v8f;

#define N_NODES 100000
#define N_EDGES 600000
#define IN_F    128
#define HID_F   256
#define OUT_F   128

// ---------------------------------------------------------------- cast f32 -> bf16
__global__ void k_cast_bf16(const float* __restrict__ in, __bf16* __restrict__ out, int n) {
    int i = blockIdx.x * blockDim.x + threadIdx.x;
    if (i < n) out[i] = (__bf16)in[i];
}

// ---------------------------------------------------------------- in-degree
__global__ void k_deg(const int* __restrict__ dst, float* __restrict__ deg, int e_total) {
    int e = blockIdx.x * blockDim.x + threadIdx.x;
    if (e < e_total) atomicAdd(&deg[dst[e]], 1.0f);
}

// ---------------------------------------------------------------- scatter-add x[src] -> agg[dst], F=128 (f32 source)
__global__ void k_gather_f32(const float* __restrict__ x, const int* __restrict__ src,
                             const int* __restrict__ dst, float* __restrict__ agg, int total) {
    int gid = blockIdx.x * blockDim.x + threadIdx.x;        // E * 32 threads, 4 feats each
    if (gid >= total) return;
    int e = gid >> 5;
    int c = (gid & 31) << 2;
    const float4 v = *(const float4*)(x + (size_t)src[e] * IN_F + c);
    float* a = agg + (size_t)dst[e] * IN_F + c;
    atomicAdd(a + 0, v.x); atomicAdd(a + 1, v.y);
    atomicAdd(a + 2, v.z); atomicAdd(a + 3, v.w);
}

// ---------------------------------------------------------------- scatter-add h1[src] -> agg[dst], F=256 (bf16 source)
__global__ void k_gather_bf16(const __bf16* __restrict__ h, const int* __restrict__ src,
                              const int* __restrict__ dst, float* __restrict__ agg, int total) {
    int gid = blockIdx.x * blockDim.x + threadIdx.x;        // E * 64 threads, 4 feats each
    if (gid >= total) return;
    int e = gid >> 6;
    int c = (gid & 63) << 2;
    v4bf v = *(const v4bf*)(h + (size_t)src[e] * HID_F + c);
    float* a = agg + (size_t)dst[e] * HID_F + c;
    atomicAdd(a + 0, (float)v[0]); atomicAdd(a + 1, (float)v[1]);
    atomicAdd(a + 2, (float)v[2]); atomicAdd(a + 3, (float)v[3]);
}

// ---------------------------------------------------------------- mean = agg / max(deg,1) -> bf16
__global__ void k_mean(const float* __restrict__ agg, const float* __restrict__ deg,
                       __bf16* __restrict__ out, int logF, int total) {
    int i = blockIdx.x * blockDim.x + threadIdx.x;
    if (i >= total) return;
    float d = deg[i >> logF];
    d = d > 1.0f ? d : 1.0f;
    out[i] = (__bf16)(agg[i] / d);
}

// ---------------------------------------------------------------- pack weights into per-lane WMMA B-fragment layout
// B tile is 32(K) x 16(N) bf16.  Lane l (<16): col n=l, K {0..7, 16..23}; lane l>=16: col n=l-16, K {8..15, 24..31}.
// Bp[((kt*NT + nt)*32 + lane)*16 + i]
__global__ void k_pack(const float* __restrict__ Wself, const float* __restrict__ Wneigh,
                       __bf16* __restrict__ Bp, int Khalf, int Nout, int total) {
    int gid = blockIdx.x * blockDim.x + threadIdx.x;
    if (gid >= total) return;
    const int NT   = Nout >> 4;
    int i    = gid & 15;
    int lane = (gid >> 4) & 31;
    int tile = gid >> 9;
    int nt   = tile % NT;
    int kt   = tile / NT;
    int kl   = ((lane >> 4) << 3) + (i < 8 ? i : i + 8);
    int k    = kt * 32 + kl;
    int n    = nt * 16 + (lane & 15);
    float v  = (k < Khalf) ? Wself[(size_t)k * Nout + n]
                           : Wneigh[(size_t)(k - Khalf) * Nout + n];
    Bp[gid] = (__bf16)v;
}

// ---------------------------------------------------------------- WMMA GEMM:  out[N, NOUT] = [A0|A1] @ Bp + bias
// A0,A1: [N_NODES, KHALF] bf16 row-major (K-concatenated).  One 16x16 output tile per wave.
template <int KHALF, int NOUT, bool RELU_BF16_OUT>
__global__ __launch_bounds__(256)
void k_gemm(const __bf16* __restrict__ A0, const __bf16* __restrict__ A1,
            const __bf16* __restrict__ Bp, const float* __restrict__ bias,
            void* __restrict__ outp) {
    constexpr int NT = NOUT / 16;
    constexpr int KT = (2 * KHALF) / 32;
    const int wave = threadIdx.x >> 5;
    const int lane = threadIdx.x & 31;
    const int nt   = blockIdx.y * 8 + wave;
    if (nt >= NT) return;
    const int mt   = blockIdx.x;
    const int row  = mt * 16 + (lane & 15);   // A row for this lane
    const int hi8  = (lane >> 4) << 3;        // +8 K-offset for upper half-wave

    v8f acc = {};
#pragma unroll
    for (int kt = 0; kt < KT; ++kt) {
        const int kbase = kt * 32;
        const __bf16* Asrc = (kbase < KHALF) ? A0 : A1;
        const int kk       = (kbase < KHALF) ? kbase : (kbase - KHALF);
        const __bf16* ap = Asrc + (size_t)row * KHALF + kk + hi8;
        v8bf lo = *(const v8bf*)(ap);          // K = kk+hi8   .. +7
        v8bf hi = *(const v8bf*)(ap + 16);     // K = kk+16+hi8.. +7
        v16bf a;
#pragma unroll
        for (int i = 0; i < 8; ++i) { a[i] = lo[i]; a[i + 8] = hi[i]; }
        v16bf b = *(const v16bf*)(Bp + ((size_t)(kt * NT + nt) * 32 + lane) * 16);
        acc = __builtin_amdgcn_wmma_f32_16x16x32_bf16(
            /*neg_a=*/false, a, /*neg_b=*/false, b,
            /*c_mod=*/(short)0, acc, /*reuse_a=*/false, /*reuse_b=*/false);
    }

    const int col = nt * 16 + (lane & 15);
    const float bv = bias[col];
    const int m0 = mt * 16 + hi8;
#pragma unroll
    for (int r = 0; r < 8; ++r) {
        float v = acc[r] + bv;
        size_t idx = (size_t)(m0 + r) * NOUT + col;
        if constexpr (RELU_BF16_OUT) {
            ((__bf16*)outp)[idx] = (__bf16)(v > 0.0f ? v : 0.0f);
        } else {
            ((float*)outp)[idx] = v;
        }
    }
}

// ---------------------------------------------------------------- edge scorer: wave per edge
__global__ __launch_bounds__(256)
void k_edge(const float* __restrict__ h2, const int* __restrict__ src,
            const int* __restrict__ dst, const float* __restrict__ We,
            const float* __restrict__ be, float* __restrict__ out, int e_total) {
    const int wave = threadIdx.x >> 5;
    const int lane = threadIdx.x & 31;
    const int e = blockIdx.x * 8 + wave;
    if (e >= e_total) return;
    const float* hs = h2 + (size_t)src[e] * OUT_F;
    const float* hd = h2 + (size_t)dst[e] * OUT_F;
    float s = 0.0f;
#pragma unroll
    for (int j = lane; j < OUT_F; j += 32)
        s += hs[j] * We[j] + hd[j] * We[OUT_F + j];
#pragma unroll
    for (int off = 16; off > 0; off >>= 1)
        s += __shfl_down(s, off, 32);
    if (lane == 0) out[e] = s + be[0];
}

// ================================================================ launch
extern "C" void kernel_launch(void* const* d_in, const int* in_sizes, int n_in,
                              void* d_out, int out_size, void* d_ws, size_t ws_size,
                              hipStream_t stream) {
    (void)in_sizes; (void)n_in; (void)out_size; (void)ws_size;
    const float* x       = (const float*)d_in[0];
    const int*   src     = (const int*)  d_in[1];
    const int*   dst     = (const int*)  d_in[2];
    const float* Wself1  = (const float*)d_in[3];
    const float* Wneigh1 = (const float*)d_in[4];
    const float* b1      = (const float*)d_in[5];
    const float* Wself2  = (const float*)d_in[6];
    const float* Wneigh2 = (const float*)d_in[7];
    const float* b2      = (const float*)d_in[8];
    const float* Wedge   = (const float*)d_in[9];
    const float* bedge   = (const float*)d_in[10];
    float* out = (float*)d_out;

    char* ws = (char*)d_ws;
    size_t off = 0;
    auto take = [&](size_t bytes) -> char* {
        char* p = ws + off;
        off = (off + bytes + 255) & ~(size_t)255;
        return p;
    };
    float*  deg    = (float*) take((size_t)N_NODES * 4);             // 0.4 MB
    float*  agg    = (float*) take((size_t)N_NODES * HID_F * 4);     // 102.4 MB (L2-resident)
    __bf16* xb     = (__bf16*)take((size_t)N_NODES * IN_F * 2);      // 25.6 MB
    __bf16* mean1b = (__bf16*)take((size_t)N_NODES * IN_F * 2);      // 25.6 MB (contiguous with xb)
    __bf16* h1b    = (__bf16*)take((size_t)N_NODES * HID_F * 2);     // 51.2 MB
    __bf16* W1p    = (__bf16*)take((size_t)2 * IN_F * HID_F * 2);
    __bf16* W2p    = (__bf16*)take((size_t)2 * HID_F * OUT_F * 2);
    __bf16* mean2b = xb;    // alias: xb+mean1b region == N*256 bf16, both dead after gemm1
    float*  h2     = agg;   // alias: agg dead after mean2

    // ---- layer 1 ----
    hipMemsetAsync(deg, 0, (size_t)N_NODES * 4, stream);
    hipMemsetAsync(agg, 0, (size_t)N_NODES * IN_F * 4, stream);
    k_cast_bf16<<<(N_NODES * IN_F) / 256, 256, 0, stream>>>(x, xb, N_NODES * IN_F);
    k_pack<<<256, 256, 0, stream>>>(Wself1, Wneigh1, W1p, IN_F, HID_F, 2 * IN_F * HID_F);
    k_pack<<<256, 256, 0, stream>>>(Wself2, Wneigh2, W2p, HID_F, OUT_F, 2 * HID_F * OUT_F);
    k_deg<<<(N_EDGES + 255) / 256, 256, 0, stream>>>(dst, deg, N_EDGES);
    k_gather_f32<<<(N_EDGES * 32) / 256, 256, 0, stream>>>(x, src, dst, agg, N_EDGES * 32);
    k_mean<<<(N_NODES * IN_F) / 256, 256, 0, stream>>>(agg, deg, mean1b, 7, N_NODES * IN_F);
    k_gemm<IN_F, HID_F, true><<<dim3(N_NODES / 16, 2), 256, 0, stream>>>(xb, mean1b, W1p, b1, h1b);

    // ---- layer 2 ----
    hipMemsetAsync(agg, 0, (size_t)N_NODES * HID_F * 4, stream);
    k_gather_bf16<<<(N_EDGES * 64) / 256, 256, 0, stream>>>(h1b, src, dst, agg, N_EDGES * 64);
    k_mean<<<(N_NODES * HID_F) / 256, 256, 0, stream>>>(agg, deg, mean2b, 8, N_NODES * HID_F);
    k_gemm<HID_F, OUT_F, false><<<dim3(N_NODES / 16, 1), 256, 0, stream>>>(h1b, mean2b, W2p, b2, h2);

    // ---- edge scores ----
    k_edge<<<N_EDGES / 8, 256, 0, stream>>>(h2, src, dst, Wedge, bedge, out, N_EDGES);
}
